// contrast_loss_58394375356785
// MI455X (gfx1250) — matrix operand order repaired and checked by reading, not compiled, not run on hardware
//
#include <hip/hip_runtime.h>
#include <hip/hip_bf16.h>

typedef _Float16 half8  __attribute__((ext_vector_type(8)));
typedef _Float16 v16h_t __attribute__((ext_vector_type(16)));
typedef float    v8f_t  __attribute__((ext_vector_type(8)));
typedef int      v4i_vs __attribute__((vector_size(16)));   // matches builtin prototype

#define N_ROWS 8192
#define DIM    256
#define EPS_N  1e-8f
// sqrt(10 * log2(e)): folds 1/temperature AND the exp->exp2 conversion into
// the normalized vectors, so sim tiles come out of WMMA ready for v_exp_f32.
#define SCALE_F 3.7982825f
#define CSPLIT 16           // column-slice parallelism
#define ROWS_PER_BLOCK 256  // 8 waves * 32 rows
#define LDS_STRIDE 264      // 256+8 halves: 528B = 4 DW mod 64 banks -> conflict-free b128

#if __has_builtin(__builtin_amdgcn_global_load_async_to_lds_b128)
#define ASYNC_COPY 1
typedef __attribute__((address_space(1))) v4i_vs* g_v4i_p;
typedef __attribute__((address_space(3))) v4i_vs* l_v4i_p;
#endif

// ---------------------------------------------------------------------------
// Kernel 0: zero the per-row accumulators and the scalar output.
// ---------------------------------------------------------------------------
__global__ void init_ws_k(float* __restrict__ denom, float* __restrict__ pos,
                          float* __restrict__ out) {
    int i = blockIdx.x * blockDim.x + threadIdx.x;
    if (i < N_ROWS) { denom[i] = 0.0f; pos[i] = 0.0f; }
    if (i == 0) out[0] = 0.0f;
}

// ---------------------------------------------------------------------------
// Kernel 1: row L2-normalize (scaled by SCALE_F), fp32 -> f16. Wave per row.
// ---------------------------------------------------------------------------
__global__ void __launch_bounds__(256)
normalize_rows_k(const float* __restrict__ y, _Float16* __restrict__ yn) {
    int gwave = (blockIdx.x * blockDim.x + threadIdx.x) >> 5;  // row index
    int lane  = threadIdx.x & 31;
    const float4* row = (const float4*)(y + (size_t)gwave * DIM);
    float4 a = row[2 * lane];
    float4 b = row[2 * lane + 1];
    float s = a.x*a.x + a.y*a.y + a.z*a.z + a.w*a.w
            + b.x*b.x + b.y*b.y + b.z*b.z + b.w*b.w;
#pragma unroll
    for (int m = 16; m >= 1; m >>= 1) s += __shfl_xor(s, m, 32);
    float inv = SCALE_F / fmaxf(sqrtf(s), EPS_N);
    half8 h;
    h[0] = (_Float16)(a.x * inv); h[1] = (_Float16)(a.y * inv);
    h[2] = (_Float16)(a.z * inv); h[3] = (_Float16)(a.w * inv);
    h[4] = (_Float16)(b.x * inv); h[5] = (_Float16)(b.y * inv);
    h[6] = (_Float16)(b.z * inv); h[7] = (_Float16)(b.w * inv);
    *(half8*)(yn + (size_t)gwave * DIM + lane * 8) = h;
}

// ---------------------------------------------------------------------------
// Kernel 2: fused Gram + exp + row-sum. Block = 8 waves; each wave owns TWO
// 16-row tiles (32 rows). B column tiles are staged into double-buffered LDS
// via async global->LDS (ASYNCcnt) and shared by all 8 waves. Diagonal /
// positive-pair handling is a wave-uniform cold branch: the hot epilogue is
// 16x (v_exp_f32 + v_add_f32) per tile.
// ---------------------------------------------------------------------------
__global__ void __launch_bounds__(256)
gram_partial_k(const _Float16* __restrict__ yn,
               float* __restrict__ denom, float* __restrict__ pos) {
    __shared__ _Float16 sB[2][16][LDS_STRIDE];   // 2 x 16 rows x 256(+8 pad)

    const int tid  = threadIdx.x;
    const int lane = tid & 31;
    const int wib  = tid >> 5;                   // wave in block (0..7)
    const int m15  = lane & 15;
    const int hi   = lane >> 4;                  // 0: lanes 0-15, 1: lanes 16-31

    const int rowbase = blockIdx.x * ROWS_PER_BLOCK + wib * 32;
    const int colBeg  = blockIdx.y * (N_ROWS / CSPLIT);
    const int nTiles  = (N_ROWS / CSPLIT) / 16;  // 32 column tiles per block

    // --- B-tile staging: 16 rows x 512B from yn[n0..n0+15][:] into sB[buf] --
    auto stage = [&](int buf, int n0) {
#pragma unroll
        for (int c = 0; c < 2; ++c) {            // 512 x 16B chunks, 2/thread
            int ch = tid + c * 256;
            int r  = ch >> 5;                    // row in tile (0..15)
            int e  = (ch & 31) * 8;              // half-element offset
            const _Float16* src = yn + (size_t)(n0 + r) * DIM + e;
            _Float16*       dst = &sB[buf][r][e];
#ifdef ASYNC_COPY
            __builtin_amdgcn_global_load_async_to_lds_b128(
                (g_v4i_p)src, (l_v4i_p)dst, 0, 0);
#else
            *(half8*)dst = *(const half8*)src;
#endif
        }
    };
    auto wait_stage = [&]() {
#ifdef ASYNC_COPY
#if __has_builtin(__builtin_amdgcn_s_wait_asynccnt)
        __builtin_amdgcn_s_wait_asynccnt(0);
#else
        asm volatile("s_wait_asynccnt 0" ::: "memory");
#endif
#endif
        __syncthreads();
    };

    // --- Preload A fragments for both row tiles (2 x 16 rows x 256 K) ------
    // A 16x32 f16 layout: lane holds row M=lane&15; lanes<16 hold K offsets
    // {0..7,16..23}, lanes>=16 hold {8..15,24..31}.
    v16h_t afA[8], afB[8];
    {
        const _Float16* ap0 = yn + (size_t)(rowbase + m15) * DIM + (hi ? 8 : 0);
        const _Float16* ap1 = ap0 + 16 * DIM;
#pragma unroll
        for (int ks = 0; ks < 8; ++ks) {
            half8 lo0 = *(const half8*)(ap0 + ks * 32);
            half8 hh0 = *(const half8*)(ap0 + ks * 32 + 16);
            half8 lo1 = *(const half8*)(ap1 + ks * 32);
            half8 hh1 = *(const half8*)(ap1 + ks * 32 + 16);
            v16h_t f0, f1;
#pragma unroll
            for (int t = 0; t < 8; ++t) {
                f0[t] = lo0[t]; f0[8 + t] = hh0[t];
                f1[t] = lo1[t]; f1[8 + t] = hh1[t];
            }
            afA[ks] = f0; afB[ks] = f1;
        }
    }

    float acc0[8], acc1[8];
#pragma unroll
    for (int g = 0; g < 8; ++g) { acc0[g] = 0.f; acc1[g] = 0.f; }

    const int M0 = rowbase + (hi ? 8 : 0);       // + g => lane's C rows, tile 0
    const int M1 = M0 + 16;                      // tile 1
    const int bk = hi ? 8 : 0;                   // B frag K offset for this half

    stage(0, colBeg);
    for (int t = 0; t < nTiles; ++t) {
        const int n0 = colBeg + t * 16;
        wait_stage();                             // staged buffer (t&1) ready
        if (t + 1 < nTiles) stage((t + 1) & 1, n0 + 16);
        const int cur = t & 1;

        v8f_t c0 = {}, c1 = {};
#pragma unroll
        for (int ks = 0; ks < 8; ++ks) {
            // B 32x16 f16 layout: lane = column N (=m15); lanes<16 hold
            // K={0..7,16..23}, lanes>=16 hold K={8..15,24..31} of this k-step.
            const _Float16* p = &sB[cur][m15][ks * 32 + bk];
            half8 b0 = *(const half8*)p;
            half8 b1 = *(const half8*)(p + 16);
            v16h_t bf;
#pragma unroll
            for (int u = 0; u < 8; ++u) { bf[u] = b0[u]; bf[8 + u] = b1[u]; }
            c0 = __builtin_amdgcn_wmma_f32_16x16x32_f16(false, afA[ks], false, bf,
                                                        (short)0, c0, false, false);
            c1 = __builtin_amdgcn_wmma_f32_16x16x32_f16(false, afB[ks], false, bf,
                                                        (short)0, c1, false, false);
        }

        // C/D layout: VGPR g, lane l -> (M = Mbase+g, N = n0 + (l&15)).
        // c already equals sim * 10 * log2(e)  =>  e = exp2(c).
        const int Ncol = n0 + m15;
        if (n0 == rowbase) {                      // wave-uniform: diag tile of c0
#pragma unroll
            for (int g = 0; g < 8; ++g) {
                float e  = __builtin_amdgcn_exp2f(c0[g]);
                int  Mg  = M0 + g;
                acc0[g] += (Ncol != Mg) ? e : 0.0f;          // exclude diagonal
                if (Ncol == (Mg ^ 1)) atomicAdd(&pos[Mg], e); // positive pair (1 lane)
            }
        } else {
#pragma unroll
            for (int g = 0; g < 8; ++g) acc0[g] += __builtin_amdgcn_exp2f(c0[g]);
        }
        if (n0 == rowbase + 16) {                 // wave-uniform: diag tile of c1
#pragma unroll
            for (int g = 0; g < 8; ++g) {
                float e  = __builtin_amdgcn_exp2f(c1[g]);
                int  Mg  = M1 + g;
                acc1[g] += (Ncol != Mg) ? e : 0.0f;
                if (Ncol == (Mg ^ 1)) atomicAdd(&pos[Mg], e);
            }
        } else {
#pragma unroll
            for (int g = 0; g < 8; ++g) acc1[g] += __builtin_amdgcn_exp2f(c1[g]);
        }
    }

    // Row-sum: reduce across the 16 lanes sharing each C row.
#pragma unroll
    for (int g = 0; g < 8; ++g) {
        float a0 = acc0[g], a1 = acc1[g];
#pragma unroll
        for (int m = 8; m >= 1; m >>= 1) {
            a0 += __shfl_xor(a0, m, 32);
            a1 += __shfl_xor(a1, m, 32);
        }
        if (m15 == 0) {
            atomicAdd(&denom[M0 + g], a0);
            atomicAdd(&denom[M1 + g], a1);
        }
    }
}

// ---------------------------------------------------------------------------
// Kernel 3: loss_i = log(denom_i) - log(pos_i); mean via atomics.
// ---------------------------------------------------------------------------
__global__ void __launch_bounds__(256)
finalize_k(const float* __restrict__ denom, const float* __restrict__ pos,
           float* __restrict__ out) {
    int i = blockIdx.x * blockDim.x + threadIdx.x;
    float l = __logf(denom[i]) - __logf(pos[i]);
#pragma unroll
    for (int m = 16; m >= 1; m >>= 1) l += __shfl_xor(l, m, 32);
    if ((threadIdx.x & 31) == 0) atomicAdd(out, l * (1.0f / (float)N_ROWS));
}

// ---------------------------------------------------------------------------
extern "C" void kernel_launch(void* const* d_in, const int* in_sizes, int n_in,
                              void* d_out, int out_size, void* d_ws, size_t ws_size,
                              hipStream_t stream) {
    const float* y   = (const float*)d_in[0];   // (8192, 256) fp32
    float*       out = (float*)d_out;           // scalar fp32

    // Workspace layout: [ yn f16: 4 MB | denom: N*4 | pos: N*4 ]
    _Float16* yn    = (_Float16*)d_ws;
    float*    denom = (float*)((char*)d_ws + (size_t)N_ROWS * DIM * sizeof(_Float16));
    float*    pos   = denom + N_ROWS;

    init_ws_k<<<(N_ROWS + 255) / 256, 256, 0, stream>>>(denom, pos, out);

    // 1 wave per row, 8 waves per block -> 1024 blocks
    normalize_rows_k<<<N_ROWS / 8, 256, 0, stream>>>(y, yn);

    // 32 row-blocks x 16 column slices = 512 blocks
    dim3 grid(N_ROWS / ROWS_PER_BLOCK, CSPLIT);
    gram_partial_k<<<grid, 256, 0, stream>>>(yn, denom, pos);

    finalize_k<<<N_ROWS / 256, 256, 0, stream>>>(denom, pos, out);
}
